// PKA_GNN_30150670418424
// MI455X (gfx1250) — compile-verified, matrix operand order checked by reading, not run on hardware
//
#include <hip/hip_runtime.h>
#include <hip/hip_bf16.h>

#define HID     128
#define NNODES  1024
#define NEDGES  16384
#define XD      133
#define BD      14

typedef __attribute__((ext_vector_type(16))) __bf16 v16bf;
typedef __attribute__((ext_vector_type(8)))  float  v8f;

// ---------------------------------------------------------------------------
// WMMA fragment helpers (CDNA5 wave32, 16x16x32 bf16, f32 accum).
// A-frag (16x32, row-major source): lanes 0-15 row=lane hold K {0..7,16..23},
// lanes 16-31 row=lane-16 hold K {8..15,24..31}.
// B-frags are pre-swizzled at pack time into fragment order:
//   tile (kt,nt) base = (kt*8 + nt)*512 elems, lane l's 16 bf16 at +l*16
// -> one aligned 32B vector load per lane (2x global_load_b128).
// C/D (16x16 f32): vgpr v, lane l -> row v + 8*(l>>4), col l&15.
// ---------------------------------------------------------------------------
__device__ __forceinline__ v16bf frag_a_lds(const __bf16* s, int ldk, int k0, int lane) {
  const int r = lane & 15, hi = (lane >> 4) & 1;
  const __bf16* p = s + r * ldk + k0 + hi * 8;
  v16bf a;
#pragma unroll
  for (int e = 0; e < 8; ++e) a[e] = p[e];
#pragma unroll
  for (int e = 0; e < 8; ++e) a[8 + e] = p[16 + e];
  return a;
}

__device__ __forceinline__ v16bf frag_b_pk(const __bf16* Wp, int kt, int nt, int lane) {
  return *(const v16bf*)(Wp + ((size_t)(kt * 8 + nt) * 512) + lane * 16);
}

__device__ __forceinline__ v8f wmma_bf16(v16bf a, v16bf b, v8f c) {
  return __builtin_amdgcn_wmma_f32_16x16x32_bf16(false, a, false, b, (short)0, c,
                                                 false, false);
}

// ---------------------------------------------------------------------------
// Weight packing: fp32 -> bf16 fragment-ordered tiles, K zero-padded to x32.
// Regions (elems): Wi 160x128 | Wh 128x128 | Wnt 160x128 | Wo 288x128 | Wg 128x128
// ---------------------------------------------------------------------------
#define PK_WI  (160 * 128)
#define PK_WH  (128 * 128)
#define PK_WNT (160 * 128)
#define PK_WO  (288 * 128)
#define PK_WG  (128 * 128)
#define PK_TOT (PK_WI + PK_WH + PK_WNT + PK_WO + PK_WG)

__device__ __forceinline__ float pick_swz(const float* W, int j, int korig) {
  // j = linear index within a fragment-ordered region of [Kpad x 128]
  const int tile = j >> 9;          // /512
  const int r    = j & 511;
  const int lane = r >> 4, e = r & 15;
  const int kt = tile >> 3, nt = tile & 7;
  const int c = lane & 15, hi = (lane >> 4) & 1;
  const int ke = (e < 8) ? (hi * 8 + e) : (16 + hi * 8 + (e - 8));
  const int k = kt * 32 + ke;
  const int n = nt * 16 + c;
  return (k < korig) ? W[k * 128 + n] : 0.f;
}

__global__ __launch_bounds__(256) void pack_weights(
    const float* __restrict__ Wi, const float* __restrict__ Wh,
    const float* __restrict__ Wnt, const float* __restrict__ Wo,
    const float* __restrict__ Wg, __bf16* __restrict__ P) {
  int idx = blockIdx.x * 256 + threadIdx.x;
  if (idx >= PK_TOT) return;
  const int s1 = PK_WI, s2 = s1 + PK_WH, s3 = s2 + PK_WNT, s4 = s3 + PK_WO;
  float v;
  if (idx < s1)       v = pick_swz(Wi,  idx,      147);
  else if (idx < s2)  v = pick_swz(Wh,  idx - s1, 128);
  else if (idx < s3)  v = pick_swz(Wnt, idx - s2, XD);
  else if (idx < s4)  v = pick_swz(Wo,  idx - s3, 261);
  else                v = pick_swz(Wg,  idx - s4, 128);
  P[idx] = (__bf16)v;
}

// ---------------------------------------------------------------------------
// H0 = concat(x[src], edge_attr) @ Wi + bi ; Ht = relu(H0)
// One block = 32 edges x 128 cols; 4 waves, each 2 row-tiles x 2 col-tiles,
// sharing B-fragments across row-tiles. K = 160 (padded).
// ---------------------------------------------------------------------------
__global__ __launch_bounds__(128) void edge_in_gemm(
    const float* __restrict__ x, const float* __restrict__ eattr,
    const int* __restrict__ src, const __bf16* __restrict__ Wi_p,
    const float* __restrict__ bi, float* __restrict__ H0,
    float* __restrict__ Ht) {
  __shared__ __bf16 As[32 * 160];
  const int tid = threadIdx.x;
  const int e0 = blockIdx.x * 32;
  for (int idx = tid; idx < 32 * 160; idx += 128) {
    int r = idx / 160, k = idx % 160;
    int e = e0 + r;
    float v = 0.f;
    if (k < XD)            v = x[(size_t)src[e] * XD + k];
    else if (k < XD + BD)  v = eattr[(size_t)e * BD + (k - XD)];
    As[idx] = (__bf16)v;
  }
  __builtin_prefetch(Wi_p, 0, 1);  // global_prefetch_b8
  __syncthreads();
  const int wv = tid >> 5, lane = tid & 31;
  const int nt0 = wv * 2;
  v8f a00 = {}, a01 = {}, a10 = {}, a11 = {};
#pragma unroll
  for (int kt = 0; kt < 5; ++kt) {
    v16bf f0 = frag_a_lds(As, 160, kt * 32, lane);
    v16bf f1 = frag_a_lds(As + 16 * 160, 160, kt * 32, lane);
    v16bf b0 = frag_b_pk(Wi_p, kt, nt0, lane);
    v16bf b1 = frag_b_pk(Wi_p, kt, nt0 + 1, lane);
    a00 = wmma_bf16(f0, b0, a00); a01 = wmma_bf16(f0, b1, a01);
    a10 = wmma_bf16(f1, b0, a10); a11 = wmma_bf16(f1, b1, a11);
  }
  const int hi = (lane >> 4) & 1, cl = lane & 15;
  const int c0 = nt0 * 16 + cl, c1 = (nt0 + 1) * 16 + cl;
#pragma unroll
  for (int v = 0; v < 8; ++v) {
    int r = v + hi * 8;
    size_t ea = (size_t)(e0 + r), eb = (size_t)(e0 + 16 + r);
    float h;
    h = a00[v] + bi[c0]; H0[ea * HID + c0] = h; Ht[ea * HID + c0] = fmaxf(h, 0.f);
    h = a01[v] + bi[c1]; H0[ea * HID + c1] = h; Ht[ea * HID + c1] = fmaxf(h, 0.f);
    h = a10[v] + bi[c0]; H0[eb * HID + c0] = h; Ht[eb * HID + c0] = fmaxf(h, 0.f);
    h = a11[v] + bi[c1]; H0[eb * HID + c1] = h; Ht[eb * HID + c1] = fmaxf(h, 0.f);
  }
}

// ---------------------------------------------------------------------------
// agg[dst[e]] += Ht[e]  (f32 global atomics, L2-side)
// ---------------------------------------------------------------------------
__global__ __launch_bounds__(256) void scatter_add(
    const float* __restrict__ Ht, const int* __restrict__ dst,
    float* __restrict__ agg) {
  int idx = blockIdx.x * 256 + threadIdx.x;
  if (idx >= NEDGES * HID) return;
  int e = idx >> 7, c = idx & 127;
  atomicAdd(&agg[(size_t)dst[e] * HID + c], Ht[idx]);
}

// ---------------------------------------------------------------------------
// HtOut = relu(H0 + (agg[src] - HtIn[rev]) @ Wh + bh)   (32 edges per block)
// ---------------------------------------------------------------------------
__global__ __launch_bounds__(128) void msg_gemm(
    const float* __restrict__ HtIn, const float* __restrict__ H0,
    const float* __restrict__ agg, const int* __restrict__ src,
    const int* __restrict__ rev, const __bf16* __restrict__ Wh_p,
    const float* __restrict__ bh, float* __restrict__ HtOut) {
  __shared__ __bf16 As[32 * 128];
  const int tid = threadIdx.x;
  const int e0 = blockIdx.x * 32;
  for (int idx = tid; idx < 32 * 128; idx += 128) {
    int r = idx >> 7, k = idx & 127;
    int e = e0 + r;
    float v = agg[(size_t)src[e] * HID + k] - HtIn[(size_t)rev[e] * HID + k];
    As[idx] = (__bf16)v;
  }
  __syncthreads();
  const int wv = tid >> 5, lane = tid & 31;
  const int nt0 = wv * 2;
  v8f a00 = {}, a01 = {}, a10 = {}, a11 = {};
#pragma unroll
  for (int kt = 0; kt < 4; ++kt) {
    v16bf f0 = frag_a_lds(As, HID, kt * 32, lane);
    v16bf f1 = frag_a_lds(As + 16 * HID, HID, kt * 32, lane);
    v16bf b0 = frag_b_pk(Wh_p, kt, nt0, lane);
    v16bf b1 = frag_b_pk(Wh_p, kt, nt0 + 1, lane);
    a00 = wmma_bf16(f0, b0, a00); a01 = wmma_bf16(f0, b1, a01);
    a10 = wmma_bf16(f1, b0, a10); a11 = wmma_bf16(f1, b1, a11);
  }
  const int hi = (lane >> 4) & 1, cl = lane & 15;
  const int c0 = nt0 * 16 + cl, c1 = (nt0 + 1) * 16 + cl;
#pragma unroll
  for (int v = 0; v < 8; ++v) {
    int r = v + hi * 8;
    size_t ea = (size_t)(e0 + r), eb = (size_t)(e0 + 16 + r);
    HtOut[ea * HID + c0] = fmaxf(H0[ea * HID + c0] + a00[v] + bh[c0], 0.f);
    HtOut[ea * HID + c1] = fmaxf(H0[ea * HID + c1] + a01[v] + bh[c1], 0.f);
    HtOut[eb * HID + c0] = fmaxf(H0[eb * HID + c0] + a10[v] + bh[c0], 0.f);
    HtOut[eb * HID + c1] = fmaxf(H0[eb * HID + c1] + a11[v] + bh[c1], 0.f);
  }
}

// ---------------------------------------------------------------------------
// tx = x@Wnt+bnt ; M = rowsum(agg)==0 ? tx : agg
// Emb = relu(concat(x, M) @ Wo + bo)   (one block = 16 nodes)
// ---------------------------------------------------------------------------
__global__ __launch_bounds__(128) void node_embed(
    const float* __restrict__ x, const float* __restrict__ agg,
    const __bf16* __restrict__ Wnt_p, const float* __restrict__ bnt,
    const __bf16* __restrict__ Wo_p, const float* __restrict__ bo,
    float* __restrict__ Emb) {
  __shared__ __bf16 Cs[16 * 288];
  __shared__ float zflag[16];
  const int tid = threadIdx.x;
  const int v0 = blockIdx.x * 16;
  for (int idx = tid; idx < 16 * 288; idx += 128) {
    int r = idx / 288, k = idx % 288;
    float v = (k < XD) ? x[(size_t)(v0 + r) * XD + k] : 0.f;
    Cs[idx] = (__bf16)v;
  }
  if (tid < 16) {
    float s = 0.f;
    for (int c = 0; c < HID; ++c) s += agg[(size_t)(v0 + tid) * HID + c];
    zflag[tid] = (s == 0.f) ? 1.f : 0.f;
  }
  __syncthreads();
  const int wv = tid >> 5, lane = tid & 31;
  const int nt0 = wv * 2;
  const int hi = (lane >> 4) & 1, cl = lane & 15;
  const int c0 = nt0 * 16 + cl, c1 = (nt0 + 1) * 16 + cl;
  // pass 1: tx = x @ Wnt  (cols 133..159 of Cs are zero, Wnt_p zero-padded)
  v8f t0 = {}, t1 = {};
#pragma unroll
  for (int kt = 0; kt < 5; ++kt) {
    v16bf a = frag_a_lds(Cs, 288, kt * 32, lane);
    t0 = wmma_bf16(a, frag_b_pk(Wnt_p, kt, nt0, lane), t0);
    t1 = wmma_bf16(a, frag_b_pk(Wnt_p, kt, nt0 + 1, lane), t1);
  }
#pragma unroll
  for (int v = 0; v < 8; ++v) {
    int r = v + hi * 8;
    float a0 = agg[(size_t)(v0 + r) * HID + c0];
    float a1 = agg[(size_t)(v0 + r) * HID + c1];
    float m0 = (zflag[r] != 0.f) ? (t0[v] + bnt[c0]) : a0;
    float m1 = (zflag[r] != 0.f) ? (t1[v] + bnt[c1]) : a1;
    Cs[r * 288 + XD + c0] = (__bf16)m0;
    Cs[r * 288 + XD + c1] = (__bf16)m1;
  }
  __syncthreads();
  // pass 2: Emb = relu(concat(x, M) @ Wo + bo), K = 288
  v8f o0 = {}, o1 = {};
#pragma unroll
  for (int kt = 0; kt < 9; ++kt) {
    v16bf a = frag_a_lds(Cs, 288, kt * 32, lane);
    o0 = wmma_bf16(a, frag_b_pk(Wo_p, kt, nt0, lane), o0);
    o1 = wmma_bf16(a, frag_b_pk(Wo_p, kt, nt0 + 1, lane), o1);
  }
#pragma unroll
  for (int v = 0; v < 8; ++v) {
    int r = v + hi * 8;
    Emb[(size_t)(v0 + r) * HID + c0] = fmaxf(o0[v] + bo[c0], 0.f);
    Emb[(size_t)(v0 + r) * HID + c1] = fmaxf(o1[v] + bo[c1], 0.f);
  }
}

// ---------------------------------------------------------------------------
// Serialized per-node loop (replicates lax.scan): one persistent workgroup,
// 256 threads = 8 waves. Gate GEMM (Emb @ Wg, 1024x128x128) via WMMA.
// ---------------------------------------------------------------------------
__global__ __launch_bounds__(256) void serial_loop(
    float* __restrict__ Emb, float* __restrict__ G,
    const __bf16* __restrict__ Wg_p, const float* __restrict__ bg,
    const float* __restrict__ Wc1, const float* __restrict__ bc1,
    const float* __restrict__ Wc2, const float* __restrict__ bc2,
    const float* __restrict__ Wc3, const float* __restrict__ bc3,
    const float* __restrict__ Wr1, const float* __restrict__ br1,
    const float* __restrict__ Wr2, const float* __restrict__ br2,
    const float* __restrict__ Wr3, const float* __restrict__ br3,
    const float* __restrict__ pka, const int* __restrict__ mask,
    float* __restrict__ out) {
  __shared__ float e_sh[HID];
  __shared__ float h1[256];
  __shared__ float h2[64];
  __shared__ float red2[2];
  __shared__ float scal[3];  // cls_sum, reg_sum, correct
  __shared__ int fire_sh;
  const int tid = threadIdx.x;
  if (tid == 0) { scal[0] = 0.f; scal[1] = 0.f; scal[2] = 0.f; }
  __syncthreads();

  for (int i = 0; i < NNODES; ++i) {
    if (tid < HID) e_sh[tid] = Emb[(size_t)i * HID + tid];
    __syncthreads();
    // classifier: 128 -> 256 -> 64 -> 2
    {
      float s = bc1[tid];
      for (int k = 0; k < HID; ++k) s = fmaf(e_sh[k], Wc1[k * 256 + tid], s);
      h1[tid] = fmaxf(s, 0.f);
    }
    __syncthreads();
    if (tid < 64) {
      float s = bc2[tid];
      for (int k = 0; k < 256; ++k) s = fmaf(h1[k], Wc2[k * 64 + tid], s);
      h2[tid] = fmaxf(s, 0.f);
    }
    __syncthreads();
    if (tid < 2) {
      float s = bc3[tid];
      for (int k = 0; k < 64; ++k) s = fmaf(h2[k], Wc3[k * 2 + tid], s);
      red2[tid] = s;
    }
    __syncthreads();
    // regressor: 128 -> 256 -> 64 -> 1 (reuse h1/h2)
    {
      float s = br1[tid];
      for (int k = 0; k < HID; ++k) s = fmaf(e_sh[k], Wr1[k * 256 + tid], s);
      h1[tid] = fmaxf(s, 0.f);
    }
    __syncthreads();
    if (tid < 64) {
      float s = br2[tid];
      for (int k = 0; k < 256; ++k) s = fmaf(h1[k], Wr2[k * 64 + tid], s);
      h2[tid] = fmaxf(s, 0.f);
    }
    __syncthreads();
    if (tid == 0) {
      float s = br3[0];
      for (int k = 0; k < 64; ++k) s = fmaf(h2[k], Wr3[k], s);
      float p = 1.f / (1.f + expf(-s));
      p = fminf(fmaxf(p, 0.f), 14.f);
      float l0 = red2[0], l1 = red2[1];
      int label = (l1 > l0) ? 1 : 0;                 // argmax, ties -> 0
      int target = (mask[i] > 0) ? 1 : 0;
      float mx = fmaxf(l0, l1);
      float lse = mx + logf(expf(l0 - mx) + expf(l1 - mx));
      scal[0] += lse - (target ? l1 : l0);           // cross entropy
      scal[2] += (label == target) ? 1.f : 0.f;
      float pv = pka[i];
      int valid = !(pv != pv);                       // !isnan
      float pvs = valid ? pv : 0.f;
      int reg_cond = (label == 1) || (mask[i] > 0);
      int fire = reg_cond && valid;
      if (fire) { float d = p - pvs; scal[1] += d * d; }
      out[i] = (float)label;
      out[NNODES + i] = reg_cond ? p : 0.f;
      fire_sh = fire;
    }
    __syncthreads();
    if (fire_sh) {  // block-uniform branch: EXEC all-ones for WMMA
      const int wv = tid >> 5, lane = tid & 31;
      const int cl = lane & 15, hi = (lane >> 4) & 1;
      for (int t = wv; t < 64 * 8; t += 8) {         // 512 16x16 tiles
        const int rb = t >> 3, nt = t & 7;
        v8f acc = {};
#pragma unroll
        for (int kt = 0; kt < 4; ++kt) {
          const float* p0 = Emb + (size_t)(rb * 16 + cl) * HID + kt * 32 + hi * 8;
          v16bf a;
#pragma unroll
          for (int e = 0; e < 8; ++e) a[e] = (__bf16)p0[e];
#pragma unroll
          for (int e = 0; e < 8; ++e) a[8 + e] = (__bf16)p0[16 + e];
          acc = wmma_bf16(a, frag_b_pk(Wg_p, kt, nt, lane), acc);
        }
#pragma unroll
        for (int v = 0; v < 8; ++v) {
          int r = rb * 16 + v + hi * 8;
          G[(size_t)r * HID + nt * 16 + cl] = acc[v];
        }
      }
      __syncthreads();
      for (int idx = tid; idx < NNODES * HID; idx += 256) {
        float g = 1.f / (1.f + expf(-(G[idx] + bg[idx & 127])));
        Emb[idx] *= (1.f - g);
      }
      __syncthreads();
    }
    __syncthreads();
  }
  if (tid == 0) {
    out[2 * NNODES + 0] = scal[0] + scal[1];
    out[2 * NNODES + 1] = scal[0];
    out[2 * NNODES + 2] = scal[1];
    out[2 * NNODES + 3] = scal[2] / (float)NNODES;
  }
}

// ---------------------------------------------------------------------------
// Workspace layout
// ---------------------------------------------------------------------------
static constexpr size_t SZ_EDGE = (size_t)NEDGES * HID * sizeof(float);  // 8 MB
static constexpr size_t SZ_NODE = (size_t)NNODES * HID * sizeof(float);  // 512 KB
static constexpr size_t OFF_H0  = 0;
static constexpr size_t OFF_HTA = OFF_H0 + SZ_EDGE;
static constexpr size_t OFF_HTB = OFF_HTA + SZ_EDGE;
static constexpr size_t OFF_AGG = OFF_HTB + SZ_EDGE;
static constexpr size_t OFF_EMB = OFF_AGG + SZ_NODE;
static constexpr size_t OFF_G   = OFF_EMB + SZ_NODE;
static constexpr size_t OFF_WP  = OFF_G + SZ_NODE;

extern "C" void kernel_launch(void* const* d_in, const int* in_sizes, int n_in,
                              void* d_out, int out_size, void* d_ws, size_t ws_size,
                              hipStream_t stream) {
  (void)in_sizes; (void)n_in; (void)out_size; (void)ws_size;
  const float* x     = (const float*)d_in[0];
  const float* eattr = (const float*)d_in[1];
  const float* pka   = (const float*)d_in[2];
  const int*   mask  = (const int*)d_in[3];
  const int*   eidx  = (const int*)d_in[4];
  const int*   rev   = (const int*)d_in[5];
  const float* Wi  = (const float*)d_in[6];  const float* bi  = (const float*)d_in[7];
  const float* Wh  = (const float*)d_in[8];  const float* bh  = (const float*)d_in[9];
  const float* Wo  = (const float*)d_in[10]; const float* bo  = (const float*)d_in[11];
  const float* Wnt = (const float*)d_in[12]; const float* bnt = (const float*)d_in[13];
  const float* Wc1 = (const float*)d_in[14]; const float* bc1 = (const float*)d_in[15];
  const float* Wc2 = (const float*)d_in[16]; const float* bc2 = (const float*)d_in[17];
  const float* Wc3 = (const float*)d_in[18]; const float* bc3 = (const float*)d_in[19];
  const float* Wr1 = (const float*)d_in[20]; const float* br1 = (const float*)d_in[21];
  const float* Wr2 = (const float*)d_in[22]; const float* br2 = (const float*)d_in[23];
  const float* Wr3 = (const float*)d_in[24]; const float* br3 = (const float*)d_in[25];
  const float* Wg  = (const float*)d_in[26]; const float* bg  = (const float*)d_in[27];

  const int* src = eidx;           // edge_index row 0
  const int* dst = eidx + NEDGES;  // edge_index row 1

  char* ws = (char*)d_ws;
  float* H0   = (float*)(ws + OFF_H0);
  float* HtA  = (float*)(ws + OFF_HTA);
  float* HtB  = (float*)(ws + OFF_HTB);
  float* agg  = (float*)(ws + OFF_AGG);
  float* Emb  = (float*)(ws + OFF_EMB);
  float* G    = (float*)(ws + OFF_G);
  __bf16* WP  = (__bf16*)(ws + OFF_WP);
  __bf16* Wi_p  = WP;
  __bf16* Wh_p  = Wi_p + PK_WI;
  __bf16* Wnt_p = Wh_p + PK_WH;
  __bf16* Wo_p  = Wnt_p + PK_WNT;
  __bf16* Wg_p  = Wo_p + PK_WO;

  pack_weights<<<(PK_TOT + 255) / 256, 256, 0, stream>>>(Wi, Wh, Wnt, Wo, Wg, WP);

  edge_in_gemm<<<NEDGES / 32, 128, 0, stream>>>(x, eattr, src, Wi_p, bi, H0, HtA);

  float* Hin = HtA;
  float* Hout = HtB;
  for (int d = 0; d < 2; ++d) {  // depth - 1 message-passing rounds
    hipMemsetAsync(agg, 0, SZ_NODE, stream);
    scatter_add<<<(NEDGES * HID) / 256, 256, 0, stream>>>(Hin, dst, agg);
    msg_gemm<<<NEDGES / 32, 128, 0, stream>>>(Hin, H0, agg, src, rev, Wh_p, bh, Hout);
    float* t = Hin; Hin = Hout; Hout = t;
  }
  hipMemsetAsync(agg, 0, SZ_NODE, stream);
  scatter_add<<<(NEDGES * HID) / 256, 256, 0, stream>>>(Hin, dst, agg);

  node_embed<<<NNODES / 16, 128, 0, stream>>>(x, agg, Wnt_p, bnt, Wo_p, bo, Emb);

  serial_loop<<<1, 256, 0, stream>>>(Emb, G, Wg_p, bg,
                                     Wc1, bc1, Wc2, bc2, Wc3, bc3,
                                     Wr1, br1, Wr2, br2, Wr3, br3,
                                     pka, mask, (float*)d_out);
}
